// Model_60438779789641
// MI455X (gfx1250) — matrix operand order repaired and checked by reading, not compiled, not run on hardware
//
#include <hip/hip_runtime.h>
#include <stdint.h>

// ---------------------------------------------------------------------------
// Problem constants
// ---------------------------------------------------------------------------
#define BN_   16384   // batch
#define CN_   1024    // channels
#define ICN_  256     // inner channels
#define CDN_  48      // coord dim
#define EPSN_ 1e-5f

typedef __bf16 v16bf __attribute__((ext_vector_type(16)));
typedef float  v8f   __attribute__((ext_vector_type(8)));
typedef int    v4i_  __attribute__((ext_vector_type(4)));

union ABFrag { v16bf v; uint4 q[2]; uint16_t s[16]; };

// Async global->LDS staging (gfx1250 ASYNCcnt path), guarded so we degrade to
// the load+ds_store path if the builtins are not declared by this toolchain.
#if defined(__has_builtin)
#if __has_builtin(__builtin_amdgcn_global_load_async_to_lds_b128) && \
    __has_builtin(__builtin_amdgcn_s_wait_asynccnt)
#define USE_ASYNC_LDS 1
#endif
#endif

#if defined(USE_ASYNC_LDS)
typedef __attribute__((address_space(1))) v4i_* as1_v4i_ptr;   // global int4*
typedef __attribute__((address_space(3))) v4i_* as3_v4i_ptr;   // LDS int4*
#endif

__device__ __forceinline__ uint16_t f32_to_bf16(float f) {
  union { float f; uint32_t u; } x; x.f = f;
  uint32_t r = x.u + 0x7FFFu + ((x.u >> 16) & 1u);   // round-to-nearest-even
  return (uint16_t)(r >> 16);
}
__device__ __forceinline__ float bf16_to_f32(uint16_t h) {
  union { uint32_t u; float f; } x; x.u = ((uint32_t)h) << 16;
  return x.f;
}
__device__ __forceinline__ float sigmoidf_(float x) {
  return 1.0f / (1.0f + __expf(-x));
}

// Stage 64 bytes (32 bf16) per thread into an LDS tile.
__device__ __forceinline__ void stage_tile(const uint16_t* __restrict__ gsrc,
                                           uint16_t* __restrict__ ldst)
{
#if defined(USE_ASYNC_LDS)
#pragma unroll
  for (int j = 0; j < 4; ++j)
    __builtin_amdgcn_global_load_async_to_lds_b128(
        (as1_v4i_ptr)(gsrc + j * 8),
        (as3_v4i_ptr)(ldst + j * 8),
        0, 0);
#else
  const uint4* s = (const uint4*)gsrc;
  uint4* d = (uint4*)ldst;
  d[0] = s[0]; d[1] = s[1]; d[2] = s[2]; d[3] = s[3];
#endif
}
__device__ __forceinline__ void wait_stage() {
#if defined(USE_ASYNC_LDS)
  __builtin_amdgcn_s_wait_asynccnt(0);
#endif
}

// ---------------------------------------------------------------------------
// K0: fold BatchNorm into linear weights, convert to bf16.
//     W' = W * s_row,  b' = (b - m)*s + beta,  s = g*rsqrt(v+eps)
// ---------------------------------------------------------------------------
__global__ void prep_weights_kernel(const float* __restrict__ W,
                                    const float* __restrict__ b,
                                    const float* __restrict__ g,
                                    const float* __restrict__ be,
                                    const float* __restrict__ m,
                                    const float* __restrict__ v,
                                    uint16_t* __restrict__ dW,
                                    float* __restrict__ dB,
                                    int N, int K, int hasBN)
{
  int idx = blockIdx.x * blockDim.x + threadIdx.x;
  int total = N * K;
  for (int i = idx; i < total; i += gridDim.x * blockDim.x) {
    int n = i / K;
    float s = hasBN ? g[n] * rsqrtf(v[n] + EPSN_) : 1.0f;
    dW[i] = f32_to_bf16(W[i] * s);
  }
  if (hasBN && idx < N) {
    float s = g[idx] * rsqrtf(v[idx] + EPSN_);
    dB[idx] = (b[idx] - m[idx]) * s + be[idx];
  }
}

// ---------------------------------------------------------------------------
// K1: xa[br][b,ch] = bf16( coor@cw^T + cb + name + type )
//     grid = (B/16, C/256, 2), block = 256
// ---------------------------------------------------------------------------
__global__ __launch_bounds__(256)
void build_xa_kernel(const float* __restrict__ name1, const float* __restrict__ type1,
                     const float* __restrict__ coor1,
                     const float* __restrict__ name2, const float* __restrict__ type2,
                     const float* __restrict__ coor2,
                     const float* __restrict__ cw, const float* __restrict__ cb,
                     uint16_t* __restrict__ xa)
{
  __shared__ float lco[16 * CDN_];
  const int br = blockIdx.z;
  const float* nm = br ? name2 : name1;
  const float* tp = br ? type2 : type1;
  const float* co = br ? coor2 : coor1;
  const int rbase = blockIdx.x * 16;
  const int ch = blockIdx.y * 256 + threadIdx.x;

  for (int i = threadIdx.x; i < 16 * CDN_; i += 256)
    lco[i] = co[(size_t)(rbase + i / CDN_) * CDN_ + (i % CDN_)];
  __syncthreads();

  float w[CDN_];
#pragma unroll
  for (int j = 0; j < CDN_; ++j) w[j] = cw[(size_t)ch * CDN_ + j];
  const float cbv = cb[ch];

  uint16_t* XA = xa + (size_t)br * BN_ * CN_;
  for (int rr = 0; rr < 16; ++rr) {
    const int b = rbase + rr;
    float c = cbv;
#pragma unroll
    for (int j = 0; j < CDN_; ++j) c = fmaf(lco[rr * CDN_ + j], w[j], c);
    const size_t off = (size_t)b * CN_ + ch;
    const float e = nm[off] + tp[off];
    XA[off] = f32_to_bf16(c + e);
  }
}

// ---------------------------------------------------------------------------
// WMMA fragment loaders (layouts per CDNA5 ISA §7.12.2)
// ---------------------------------------------------------------------------
__device__ __forceinline__ void load_a_frag(ABFrag& a, const uint16_t* __restrict__ Abase,
                                            int row, int ld, int k0, int lane)
{
  // lane L: M = L%16 ; e0..7 -> K = k0 + (L>=16?8:0) + 0..7 ; e8..15 -> +16
  const uint16_t* p = Abase + (size_t)row * ld + k0 + ((lane >> 4) << 3);
  a.q[0] = *(const uint4*)p;
  a.q[1] = *(const uint4*)(p + 16);
}
__device__ __forceinline__ void load_b_frag(ABFrag& b, const uint16_t* __restrict__ lds_tile,
                                            int n_local, int lane)
{
  // lane L: N = L%16 ; e0..7 -> K = (L>=16?8:0) + 0..7 ; e8..15 -> +16. LDS tile layout [n][32] bf16.
  const uint16_t* p = lds_tile + (size_t)n_local * 32 + ((lane >> 4) << 3);
  b.q[0] = *(const uint4*)p;
  b.q[1] = *(const uint4*)(p + 16);
}

// ---------------------------------------------------------------------------
// K2: h[combo] = relu( xa[br] @ W1'^T + b1' )   [B,1024] x [256,1024] -> [B,256]
//     grid = (B/128, 1, 4 combos), block = 256 (8 waves x 16 rows)
//     Double-buffered async weight tiles; B-frags batched 8-wide.
// ---------------------------------------------------------------------------
__global__ __launch_bounds__(256)
void gemm1_kernel(const uint16_t* __restrict__ xa,
                  const uint16_t* __restrict__ wA_l, const uint16_t* __restrict__ wA_g,
                  const float* __restrict__ bias_l, const float* __restrict__ bias_g,
                  uint16_t* __restrict__ hbuf)
{
  __shared__ __align__(16) uint16_t lw[2][256 * 32];   // 2 x 16 KB
  const int combo = blockIdx.z;                         // 0..3 = (br,att)
  const int br = combo >> 1, att = combo & 1;
  const uint16_t* A = xa + (size_t)br * BN_ * CN_;
  const uint16_t* W = att ? wA_g : wA_l;
  const float* bias = att ? bias_g : bias_l;
  uint16_t* H = hbuf + (size_t)combo * BN_ * ICN_;

  const int lane = threadIdx.x & 31;
  const int wave = threadIdx.x >> 5;
  const int m0 = blockIdx.x * 128 + wave * 16;
  const int arow = m0 + (lane & 15);
  const int KT = CN_ / 32;

  v8f acc[16];
  const v8f z = {0,0,0,0,0,0,0,0};
#pragma unroll
  for (int i = 0; i < 16; ++i) acc[i] = z;

  // prologue: stage tile 0
  stage_tile(W + (size_t)threadIdx.x * CN_, &lw[0][threadIdx.x * 32]);
  wait_stage();
  __syncthreads();

  for (int kt = 0; kt < KT; ++kt) {
    const int cur = kt & 1;
    if (kt + 1 < KT) {
      stage_tile(W + (size_t)threadIdx.x * CN_ + (kt + 1) * 32,
                 &lw[cur ^ 1][threadIdx.x * 32]);
      if (kt + 2 < KT)
        __builtin_prefetch(W + (size_t)threadIdx.x * CN_ + (kt + 2) * 32, 0, 3);
    }

    ABFrag a; load_a_frag(a, A, arow, CN_, kt * 32, lane);
#pragma unroll
    for (int half = 0; half < 2; ++half) {
      ABFrag bf[8];
#pragma unroll
      for (int j = 0; j < 8; ++j)
        load_b_frag(bf[j], lw[cur], (half * 8 + j) * 16 + (lane & 15), lane);
#pragma unroll
      for (int j = 0; j < 8; ++j)
        acc[half * 8 + j] = __builtin_amdgcn_wmma_f32_16x16x32_bf16(
            false, a.v, false, bf[j].v, (short)0, acc[half * 8 + j], false, false);
    }
    wait_stage();
    __syncthreads();
  }

#pragma unroll
  for (int nt = 0; nt < 16; ++nt) {
    const int n = nt * 16 + (lane & 15);
    const float bv = bias[n];
#pragma unroll
    for (int r = 0; r < 8; ++r) {
      const int row = m0 + r + ((lane >> 4) << 3);
      float val = acc[nt][r] + bv;
      val = val > 0.0f ? val : 0.0f;
      H[(size_t)row * ICN_ + n] = f32_to_bf16(val);
    }
  }
}

// ---------------------------------------------------------------------------
// K3: pre = h_l@W2l'^T + h_g@W2g'^T + b2l' + b2g';  wei = sigmoid(pre)
//     t   = 2e + 2*wei*(xa - 2e);  store bf16 into r[:, br*1024 + n]
//     grid = (B/128, 1024/256, 2), block = 256
// ---------------------------------------------------------------------------
__global__ __launch_bounds__(256)
void gemm2_combine_kernel(const uint16_t* __restrict__ hbuf,
                          const uint16_t* __restrict__ wB_l, const uint16_t* __restrict__ wB_g,
                          const float* __restrict__ bias2l, const float* __restrict__ bias2g,
                          const uint16_t* __restrict__ xa,
                          const float* __restrict__ name1, const float* __restrict__ type1,
                          const float* __restrict__ name2, const float* __restrict__ type2,
                          uint16_t* __restrict__ rws)
{
  __shared__ __align__(16) uint16_t lwl[2][256 * 32];   // 2 x 16 KB
  __shared__ __align__(16) uint16_t lwg[2][256 * 32];   // 2 x 16 KB
  const int br = blockIdx.z;
  const int nbase = blockIdx.y * 256;
  const uint16_t* Al = hbuf + (size_t)(br * 2 + 0) * BN_ * ICN_;
  const uint16_t* Ag = hbuf + (size_t)(br * 2 + 1) * BN_ * ICN_;
  const float* nm = br ? name2 : name1;
  const float* tp = br ? type2 : type1;
  const uint16_t* XA = xa + (size_t)br * BN_ * CN_;

  const int lane = threadIdx.x & 31;
  const int wave = threadIdx.x >> 5;
  const int m0 = blockIdx.x * 128 + wave * 16;
  const int arow = m0 + (lane & 15);
  const int KT = ICN_ / 32;
  const int nrow = nbase + threadIdx.x;

  v8f acc[16];
  const v8f z = {0,0,0,0,0,0,0,0};
#pragma unroll
  for (int i = 0; i < 16; ++i) acc[i] = z;

  stage_tile(wB_l + (size_t)nrow * ICN_, &lwl[0][threadIdx.x * 32]);
  stage_tile(wB_g + (size_t)nrow * ICN_, &lwg[0][threadIdx.x * 32]);
  wait_stage();
  __syncthreads();

  for (int kt = 0; kt < KT; ++kt) {
    const int cur = kt & 1;
    if (kt + 1 < KT) {
      stage_tile(wB_l + (size_t)nrow * ICN_ + (kt + 1) * 32, &lwl[cur ^ 1][threadIdx.x * 32]);
      stage_tile(wB_g + (size_t)nrow * ICN_ + (kt + 1) * 32, &lwg[cur ^ 1][threadIdx.x * 32]);
    }

    ABFrag al, ag;
    load_a_frag(al, Al, arow, ICN_, kt * 32, lane);
    load_a_frag(ag, Ag, arow, ICN_, kt * 32, lane);
#pragma unroll
    for (int grp = 0; grp < 4; ++grp) {
      ABFrag bl[4], bg[4];
#pragma unroll
      for (int j = 0; j < 4; ++j) {
        const int nl = (grp * 4 + j) * 16 + (lane & 15);
        load_b_frag(bl[j], lwl[cur], nl, lane);
        load_b_frag(bg[j], lwg[cur], nl, lane);
      }
#pragma unroll
      for (int j = 0; j < 4; ++j) {
        acc[grp * 4 + j] = __builtin_amdgcn_wmma_f32_16x16x32_bf16(
            false, al.v, false, bl[j].v, (short)0, acc[grp * 4 + j], false, false);
        acc[grp * 4 + j] = __builtin_amdgcn_wmma_f32_16x16x32_bf16(
            false, ag.v, false, bg[j].v, (short)0, acc[grp * 4 + j], false, false);
      }
    }
    wait_stage();
    __syncthreads();
  }

#pragma unroll
  for (int nt = 0; nt < 16; ++nt) {
    const int n = nbase + nt * 16 + (lane & 15);
    const float bv = bias2l[n] + bias2g[n];
#pragma unroll
    for (int r = 0; r < 8; ++r) {
      const int row = m0 + r + ((lane >> 4) << 3);
      const float pre = acc[nt][r] + bv;
      const float wei = sigmoidf_(pre);
      const size_t off = (size_t)row * CN_ + n;
      const float e = nm[off] + tp[off];
      const float xv = bf16_to_f32(XA[off]);
      // t = 2*c*wei + 2*e*(1-wei), with c = xa - e  ->  t = 2e + 2*wei*(xa - 2e)
      const float t = 2.0f * e + 2.0f * wei * (xv - 2.0f * e);
      rws[(size_t)row * (2 * CN_) + br * CN_ + n] = f32_to_bf16(t);
    }
  }
}

// ---------------------------------------------------------------------------
// K4: h2 = relu(r @ fw1^T + fb1); out[b] = sigmoid(h2[b,:].fw2 + fb2)
//     grid = (B/128), block = 256; final dot fused via shfl_xor reduction.
// ---------------------------------------------------------------------------
__global__ __launch_bounds__(256)
void gemm3_out_kernel(const uint16_t* __restrict__ rws,
                      const uint16_t* __restrict__ wF1,
                      const float* __restrict__ fb1,
                      const float* __restrict__ fw2, const float* __restrict__ fb2,
                      float* __restrict__ out)
{
  __shared__ __align__(16) uint16_t lw[2][256 * 32];
  const int lane = threadIdx.x & 31;
  const int wave = threadIdx.x >> 5;
  const int m0 = blockIdx.x * 128 + wave * 16;
  const int arow = m0 + (lane & 15);
  const int K = 2 * CN_;   // 2048
  const int KT = K / 32;

  v8f acc[16];
  const v8f z = {0,0,0,0,0,0,0,0};
#pragma unroll
  for (int i = 0; i < 16; ++i) acc[i] = z;

  stage_tile(wF1 + (size_t)threadIdx.x * K, &lw[0][threadIdx.x * 32]);
  wait_stage();
  __syncthreads();

  for (int kt = 0; kt < KT; ++kt) {
    const int cur = kt & 1;
    if (kt + 1 < KT) {
      stage_tile(wF1 + (size_t)threadIdx.x * K + (kt + 1) * 32,
                 &lw[cur ^ 1][threadIdx.x * 32]);
      if (kt + 2 < KT)
        __builtin_prefetch(wF1 + (size_t)threadIdx.x * K + (kt + 2) * 32, 0, 3);
    }

    ABFrag a; load_a_frag(a, rws, arow, K, kt * 32, lane);
#pragma unroll
    for (int half = 0; half < 2; ++half) {
      ABFrag bf[8];
#pragma unroll
      for (int j = 0; j < 8; ++j)
        load_b_frag(bf[j], lw[cur], (half * 8 + j) * 16 + (lane & 15), lane);
#pragma unroll
      for (int j = 0; j < 8; ++j)
        acc[half * 8 + j] = __builtin_amdgcn_wmma_f32_16x16x32_bf16(
            false, a.v, false, bf[j].v, (short)0, acc[half * 8 + j], false, false);
    }
    wait_stage();
    __syncthreads();
  }

  // Fused head: partial[r] = sum_n relu(acc + fb1[n]) * fw2[n]
  float partial[8];
#pragma unroll
  for (int r = 0; r < 8; ++r) partial[r] = 0.0f;
#pragma unroll
  for (int nt = 0; nt < 16; ++nt) {
    const int n = nt * 16 + (lane & 15);
    const float bv = fb1[n];
    const float w2 = fw2[n];
#pragma unroll
    for (int r = 0; r < 8; ++r) {
      float val = acc[nt][r] + bv;
      val = val > 0.0f ? val : 0.0f;
      partial[r] = fmaf(val, w2, partial[r]);
    }
  }
  // Reduce over the 16 N-lanes in each half (rows 0-7 live in lanes 0-15,
  // rows 8-15 in lanes 16-31 per the C/D fragment layout).
#pragma unroll
  for (int msk = 1; msk < 16; msk <<= 1)
#pragma unroll
    for (int r = 0; r < 8; ++r)
      partial[r] += __shfl_xor(partial[r], msk, 32);

  const float fb2v = fb2[0];
  if (lane == 0) {
#pragma unroll
    for (int r = 0; r < 8; ++r) out[m0 + r] = sigmoidf_(partial[r] + fb2v);
  }
  if (lane == 16) {
#pragma unroll
    for (int r = 0; r < 8; ++r) out[m0 + 8 + r] = sigmoidf_(partial[r] + fb2v);
  }
}

// ---------------------------------------------------------------------------
// Host launcher
// ---------------------------------------------------------------------------
extern "C" void kernel_launch(void* const* d_in, const int* in_sizes, int n_in,
                              void* d_out, int out_size, void* d_ws, size_t ws_size,
                              hipStream_t stream)
{
  const float* name1 = (const float*)d_in[0];
  const float* type1 = (const float*)d_in[1];
  const float* coor1 = (const float*)d_in[2];
  const float* name2 = (const float*)d_in[3];
  const float* type2 = (const float*)d_in[4];
  const float* coor2 = (const float*)d_in[5];
  const float* cw    = (const float*)d_in[6];
  const float* cb    = (const float*)d_in[7];
  const float* l1_w = (const float*)d_in[8];  const float* l1_b = (const float*)d_in[9];
  const float* l1_g = (const float*)d_in[10]; const float* l1_be = (const float*)d_in[11];
  const float* l1_m = (const float*)d_in[12]; const float* l1_v = (const float*)d_in[13];
  const float* l2_w = (const float*)d_in[14]; const float* l2_b = (const float*)d_in[15];
  const float* l2_g = (const float*)d_in[16]; const float* l2_be = (const float*)d_in[17];
  const float* l2_m = (const float*)d_in[18]; const float* l2_v = (const float*)d_in[19];
  const float* g1_w = (const float*)d_in[20]; const float* g1_b = (const float*)d_in[21];
  const float* g1_g = (const float*)d_in[22]; const float* g1_be = (const float*)d_in[23];
  const float* g1_m = (const float*)d_in[24]; const float* g1_v = (const float*)d_in[25];
  const float* g2_w = (const float*)d_in[26]; const float* g2_b = (const float*)d_in[27];
  const float* g2_g = (const float*)d_in[28]; const float* g2_be = (const float*)d_in[29];
  const float* g2_m = (const float*)d_in[30]; const float* g2_v = (const float*)d_in[31];
  const float* fw1 = (const float*)d_in[32]; const float* fb1 = (const float*)d_in[33];
  const float* fw2 = (const float*)d_in[34]; const float* fb2 = (const float*)d_in[35];
  (void)in_sizes; (void)n_in; (void)out_size; (void)ws_size;

  char* ws = (char*)d_ws;
  // weights (bf16) + folded biases: first 4 MB
  uint16_t* wA_l = (uint16_t*)(ws + ((size_t)0 << 10));
  uint16_t* wA_g = (uint16_t*)(ws + ((size_t)512 << 10));
  uint16_t* wB_l = (uint16_t*)(ws + ((size_t)1024 << 10));
  uint16_t* wB_g = (uint16_t*)(ws + ((size_t)1536 << 10));
  uint16_t* wF1  = (uint16_t*)(ws + ((size_t)2048 << 10));
  float* b1l = (float*)(ws + ((size_t)3072 << 10));
  float* b1g = b1l + ICN_;
  float* b2l = b1g + ICN_;
  float* b2g = b2l + CN_;
  // activations
  const size_t base = (size_t)4 << 20;
  uint16_t* xa   = (uint16_t*)(ws + base);                           // 2*B*C bf16 = 64 MB
  uint16_t* hbuf = (uint16_t*)(ws + base + ((size_t)64 << 20));      // 4*B*IC bf16 = 32 MB
  uint16_t* rws  = (uint16_t*)(ws + base + ((size_t)96 << 20));      // B*2C bf16 = 64 MB
  float* out = (float*)d_out;

  // K0: BN-fold + bf16 convert
  {
    const int blk = 256;
    int tot = ICN_ * CN_;  // 262144 (l1/g1)
    prep_weights_kernel<<<(tot + blk - 1) / blk, blk, 0, stream>>>(
        l1_w, l1_b, l1_g, l1_be, l1_m, l1_v, wA_l, b1l, ICN_, CN_, 1);
    prep_weights_kernel<<<(tot + blk - 1) / blk, blk, 0, stream>>>(
        g1_w, g1_b, g1_g, g1_be, g1_m, g1_v, wA_g, b1g, ICN_, CN_, 1);
    tot = CN_ * ICN_;      // 262144 (l2/g2)
    prep_weights_kernel<<<(tot + blk - 1) / blk, blk, 0, stream>>>(
        l2_w, l2_b, l2_g, l2_be, l2_m, l2_v, wB_l, b2l, CN_, ICN_, 1);
    prep_weights_kernel<<<(tot + blk - 1) / blk, blk, 0, stream>>>(
        g2_w, g2_b, g2_g, g2_be, g2_m, g2_v, wB_g, b2g, CN_, ICN_, 1);
    tot = ICN_ * 2 * CN_;  // 524288 (fw1)
    prep_weights_kernel<<<(tot + blk - 1) / blk, blk, 0, stream>>>(
        fw1, nullptr, nullptr, nullptr, nullptr, nullptr, wF1, nullptr, ICN_, 2 * CN_, 0);
  }

  // K1: xa = c + e  (both branches)
  build_xa_kernel<<<dim3(BN_ / 16, CN_ / 256, 2), 256, 0, stream>>>(
      name1, type1, coor1, name2, type2, coor2, cw, cb, xa);

  // K2: h[combo] = relu(BN(xa @ W1^T + b1))   (4 GEMMs)
  gemm1_kernel<<<dim3(BN_ / 128, 1, 4), 256, 0, stream>>>(
      xa, wA_l, wA_g, b1l, b1g, hbuf);

  // K3: wei = sigmoid(att_l + att_g); combine -> r (both branches)
  gemm2_combine_kernel<<<dim3(BN_ / 128, CN_ / 256, 2), 256, 0, stream>>>(
      hbuf, wB_l, wB_g, b2l, b2g, xa, name1, type1, name2, type2, rws);

  // K4: final MLP head fully fused -> out
  gemm3_out_kernel<<<dim3(BN_ / 128, 1, 1), 256, 0, stream>>>(
      rws, wF1, fb1, fw2, fb2, out);
}
